// TRUST_PATH_74431783239687
// MI455X (gfx1250) — matrix operand order repaired and checked by reading, not compiled, not run on hardware
//
#include <hip/hip_runtime.h>
#include <hip/hip_bf16.h>
#include <stdint.h>

// Problem constants (match reference)
constexpr int kB  = 128;
constexpr int kL  = 50;
constexpr int kH  = 64;
constexpr int kNH = 4;
constexpr int kNU = 200000;

typedef __attribute__((ext_vector_type(16))) __bf16 v16bf;
typedef __attribute__((ext_vector_type(8)))  float  v8f;

__device__ __forceinline__ __bf16 f2bf(float f) {
  union { float f; uint32_t u; } v; v.f = f;
  uint32_t r = v.u + 0x7FFFu + ((v.u >> 16) & 1u);   // round-to-nearest-even
  uint16_t h = (uint16_t)(r >> 16);
  return __builtin_bit_cast(__bf16, h);
}
__device__ __forceinline__ float bf2f(__bf16 b) {
  uint16_t h = __builtin_bit_cast(uint16_t, b);
  union { uint32_t u; float f; } v; v.u = ((uint32_t)h) << 16;
  return v.f;
}
__device__ __forceinline__ float elu1(float x) { return x > 0.f ? x : (__expf(x) - 1.f); }

// ---------------------------------------------------------------- gather / lengths
__global__ void k_gather(const int* __restrict__ path, const float* __restrict__ emb,
                         float* __restrict__ h) {
  int idx = blockIdx.x * blockDim.x + threadIdx.x;
  if (idx >= kB * kL * kH) return;
  int c  = idx & (kH - 1);
  int bl = idx >> 6;
  h[idx] = emb[(size_t)path[bl] * kH + c];
}

__global__ void k_seqlen(const int* __restrict__ mask, int* __restrict__ seql) {
  int b = blockIdx.x * blockDim.x + threadIdx.x;
  if (b < kB) {
    int s = 0;
    for (int l = 0; l < kL; ++l) s += mask[b * kL + l];
    seql[b] = s;
  }
}

// ---------------------------------------------------------------- GAT (one WG per head x batch)
__global__ void __launch_bounds__(256)
k_gat(const float* __restrict__ hin, const float* __restrict__ W,
      const float* __restrict__ a, const int* __restrict__ mask,
      float* __restrict__ out, int nheads, int out_stride, int do_elu) {
  __shared__ float sW[kH][kH];          // 16 KB
  __shared__ float sh[kL][kH];          // 12.8 KB
  __shared__ float sWh[kL][kH + 1];     // ~13 KB (pad vs bank conflicts)
  __shared__ float sf1[kL];
  __shared__ float sf2[kL];
  __shared__ float sP[8][kL];           // per-wave attention probs

  int tid  = threadIdx.x;
  int head = blockIdx.x % nheads;
  int b    = blockIdx.x / nheads;

  for (int i = tid; i < kH * kH; i += 256) sW[i >> 6][i & 63] = W[(size_t)head * kH * kH + i];
  for (int i = tid; i < kL * kH; i += 256) sh[i >> 6][i & 63] = hin[(size_t)b * kL * kH + i];
  __syncthreads();

  // Wh = h @ W
  for (int i = tid; i < kL * kH; i += 256) {
    int r = i >> 6, c = i & 63;
    float acc = 0.f;
    #pragma unroll 8
    for (int k = 0; k < kH; ++k) acc += sh[r][k] * sW[k][c];
    sWh[r][c] = acc;
  }
  __syncthreads();

  // f1 = Wh @ a[:H], f2 = Wh @ a[H:]
  if (tid < kL) {
    float acc = 0.f;
    for (int k = 0; k < kH; ++k) acc += sWh[tid][k] * a[head * 2 * kH + k];
    sf1[tid] = acc;
  } else if (tid >= 128 && tid < 128 + kL) {
    int r = tid - 128;
    float acc = 0.f;
    for (int k = 0; k < kH; ++k) acc += sWh[r][k] * a[head * 2 * kH + kH + k];
    sf2[r] = acc;
  }
  __syncthreads();

  // masked softmax rows + attn @ Wh ; one wave (32 lanes) per query row
  int wave = tid >> 5, lane = tid & 31;
  for (int i = wave; i < kL; i += 8) {
    float f1i = sf1[i];
    int j0 = lane, j1 = lane + 32;
    float e0 = -9.0e15f, e1 = -9.0e15f;
    if (mask[b * kL + j0] > 0)            { float x = f1i + sf2[j0]; e0 = x > 0.f ? x : 0.2f * x; }
    if (j1 < kL && mask[b * kL + j1] > 0) { float x = f1i + sf2[j1]; e1 = x > 0.f ? x : 0.2f * x; }
    float mx = fmaxf(e0, e1);
    #pragma unroll
    for (int off = 16; off > 0; off >>= 1) mx = fmaxf(mx, __shfl_xor(mx, off, 32));
    float s0 = __expf(e0 - mx);
    float s1 = (j1 < kL) ? __expf(e1 - mx) : 0.f;
    float ssum = s0 + s1;
    #pragma unroll
    for (int off = 16; off > 0; off >>= 1) ssum += __shfl_xor(ssum, off, 32);
    float inv = 1.f / ssum;
    sP[wave][j0] = s0 * inv;
    if (j1 < kL) sP[wave][j1] = s1 * inv;
    // out[i,:] = sum_j p_j * Wh[j,:]  (lane owns 2 columns)
    float acc0 = 0.f, acc1 = 0.f;
    int c0 = lane, c1 = lane + 32;
    for (int j = 0; j < kL; ++j) {
      float p = sP[wave][j];
      acc0 += p * sWh[j][c0];
      acc1 += p * sWh[j][c1];
    }
    if (do_elu) { acc0 = elu1(acc0); acc1 = elu1(acc1); }
    size_t ro = ((size_t)b * kL + i) * out_stride + (size_t)head * kH;
    out[ro + c0] = acc0;
    out[ro + c1] = acc1;
  }
}

// ---------------------------------------------------------------- mul_one = elu(mul_seq @ w)
__global__ void k_mulone(const float* __restrict__ mul_seq, const float* __restrict__ w,
                         float* __restrict__ mul_one) {
  int idx = blockIdx.x * blockDim.x + threadIdx.x;
  if (idx >= kB * kL * kH) return;
  int c = idx & 63, row = idx >> 6;
  const float* mrow = mul_seq + (size_t)row * (kNH * kH);
  float acc = 0.f;
  for (int k = 0; k < kNH * kH; ++k) acc += mrow[k] * w[k * kH + c];
  mul_one[idx] = elu1(acc);
}

// ---------------------------------------------------------------- attention readout -> a[128,64]
__global__ void __launch_bounds__(64)
k_readout(const float* __restrict__ seq_hidden, const int* __restrict__ seql,
          const int* __restrict__ mask,
          const float* __restrict__ l1W, const float* __restrict__ l1b,
          const float* __restrict__ l2W, const float* __restrict__ l2b,
          const float* __restrict__ l3W,
          const float* __restrict__ ltW, const float* __restrict__ ltb,
          float* __restrict__ afin) {
  __shared__ float sh[kL][kH];
  __shared__ float sht[kH];
  __shared__ float red[kH];
  __shared__ float salpha[kL];
  int b = blockIdx.x, c = threadIdx.x;
  for (int i = c; i < kL * kH; i += kH) sh[i >> 6][i & 63] = seq_hidden[(size_t)b * kL * kH + i];
  __syncthreads();
  int last = seql[b] - 1;
  sht[c] = sh[last][c];
  __syncthreads();
  float q1 = l1b[c];
  for (int k = 0; k < kH; ++k) q1 += sht[k] * l1W[k * kH + c];
  for (int l = 0; l < kL; ++l) {
    float q2 = l2b[c];
    for (int k = 0; k < kH; ++k) q2 += sh[l][k] * l2W[k * kH + c];
    float sg = 1.f / (1.f + __expf(-(q1 + q2)));
    red[c] = sg * l3W[c];
    __syncthreads();
    if (c == 0) { float s = 0.f; for (int k2 = 0; k2 < kH; ++k2) s += red[k2]; salpha[l] = s; }
    __syncthreads();
  }
  float s = 0.f;
  for (int l = 0; l < kL; ++l) if (mask[b * kL + l] > 0) s += salpha[l] * sh[l][c];
  red[c] = s;
  __syncthreads();
  float o = ltb[c];
  for (int k = 0; k < kH; ++k)
    o += red[k] * ltW[k * kH + c] + sht[k] * ltW[(kH + k) * kH + c];
  afin[b * kH + c] = o;
}

// ---------------------------------------------------------------- precompute A fragments (bf16 hi/lo, WMMA layout)
// a[128,64] is reused by all 12,500 score blocks: split it ONCE into the exact
// per-(M-tile, lane) operand layout so k_scores does zero conversion VALU.
//  A 16x32 (16-bit) layout: lane m = lane&15 ; K = ch*32 + hi*16 + e
__global__ void __launch_bounds__(256)
k_afrag(const float* __restrict__ afin, __bf16* __restrict__ ahfrag,
        __bf16* __restrict__ alfrag) {
  int tid  = threadIdx.x;            // (mtile, lane) = (tid>>5, tid&31)
  int mt   = tid >> 5, lane = tid & 31;
  int mloc = lane & 15, hi = lane >> 4;
  #pragma unroll
  for (int ch = 0; ch < 2; ++ch) {
    #pragma unroll
    for (int e = 0; e < 16; ++e) {
      int ka = ch * 32 + hi * 16 + e;
      float x = afin[(mt * 16 + mloc) * kH + ka];
      __bf16 xh = f2bf(x);
      ahfrag[(tid * 2 + ch) * 16 + e] = xh;
      alfrag[(tid * 2 + ch) * 16 + e] = f2bf(x - bf2f(xh));
    }
  }
}

// ---------------------------------------------------------------- scores = a @ emb[:NU]^T  (WMMA, bf16 hi/lo split)
// One block per 16-column N-tile; 8 waves each own one 16-row M-tile.
// emb tile is split to bf16 hi/lo cooperatively in LDS (once per block);
// fragments then come from 32B-aligned contiguous ds/vector loads.
__global__ void __launch_bounds__(256)
k_scores(const __bf16* __restrict__ ahfrag, const __bf16* __restrict__ alfrag,
         const float* __restrict__ emb, float* __restrict__ out) {
  __shared__ __bf16 sBh[16][kH];     // [n][k] hi parts, 2 KB
  __shared__ __bf16 sBl[16][kH];     // [n][k] lo parts, 2 KB
  int tid = threadIdx.x;
  int n0  = blockIdx.x * 16;         // 12,500 exact tiles

  // stage + split the 16x64 emb tile: 1024 floats, 4 per thread
  for (int i = tid; i < 16 * kH; i += 256) {
    int r = i >> 6, k = i & 63;
    float y = emb[(size_t)(n0 + r) * kH + k];
    __bf16 yh = f2bf(y);
    sBh[r][k] = yh;
    sBl[r][k] = f2bf(y - bf2f(yh));
  }
  // prefetch next block's tile while we compute this one (global_prefetch_b8)
  if (n0 + 16 < kNU)
    __builtin_prefetch(&emb[(size_t)(n0 + 16 + (tid >> 4)) * kH + (tid & 15) * 4], 0, 1);
  __syncthreads();

  int wave = tid >> 5, lane = tid & 31;
  int mloc = lane & 15, hi = lane >> 4;

  // A fragments: precomputed, contiguous 32 B per (wave,lane,chunk); L2-resident
  const v16bf* pah = (const v16bf*)ahfrag;
  const v16bf* pal = (const v16bf*)alfrag;
  v16bf ah0 = pah[tid * 2 + 0], ah1 = pah[tid * 2 + 1];
  v16bf al0 = pal[tid * 2 + 0], al1 = pal[tid * 2 + 1];

  // B fragments (32x16): lane n = lane&15 ; halves K = ch*32 + hi*16 + e
  // -> one contiguous, 32B-aligned 16-half run per fragment
  v16bf bh0 = *(const v16bf*)&sBh[mloc][hi * 16];
  v16bf bh1 = *(const v16bf*)&sBh[mloc][32 + hi * 16];
  v16bf bl0 = *(const v16bf*)&sBl[mloc][hi * 16];
  v16bf bl1 = *(const v16bf*)&sBl[mloc][32 + hi * 16];

  // fp32-accurate product via bf16 split: ah*bh + ah*bl + al*bh (per K-chunk)
  v8f c = {};
  c = __builtin_amdgcn_wmma_f32_16x16x32_bf16(false, ah0, false, bh0, (short)0, c, false, false);
  c = __builtin_amdgcn_wmma_f32_16x16x32_bf16(false, ah0, false, bl0, (short)0, c, false, false);
  c = __builtin_amdgcn_wmma_f32_16x16x32_bf16(false, al0, false, bh0, (short)0, c, false, false);
  c = __builtin_amdgcn_wmma_f32_16x16x32_bf16(false, ah1, false, bh1, (short)0, c, false, false);
  c = __builtin_amdgcn_wmma_f32_16x16x32_bf16(false, ah1, false, bl1, (short)0, c, false, false);
  c = __builtin_amdgcn_wmma_f32_16x16x32_bf16(false, al1, false, bh1, (short)0, c, false, false);

  // C/D layout: lanes 0-15 -> N=lane, M=v ; lanes 16-31 -> N=lane-16, M=v+8
  #pragma unroll
  for (int v = 0; v < 8; ++v)
    out[(size_t)(wave * 16 + hi * 8 + v) * kNU + (n0 + mloc)] = c[v];
}

// ---------------------------------------------------------------- launch
extern "C" void kernel_launch(void* const* d_in, const int* in_sizes, int n_in,
                              void* d_out, int out_size, void* d_ws, size_t ws_size,
                              hipStream_t stream) {
  (void)in_sizes; (void)n_in; (void)out_size; (void)ws_size;
  const int*   path  = (const int*)d_in[0];
  const int*   mask  = (const int*)d_in[1];
  const float* emb   = (const float*)d_in[2];
  const float* W_in  = (const float*)d_in[3];
  const float* a_in  = (const float*)d_in[4];
  const float* w     = (const float*)d_in[5];
  const float* W_out = (const float*)d_in[6];
  const float* a_out = (const float*)d_in[7];
  const float* l1W   = (const float*)d_in[8];
  const float* l1b   = (const float*)d_in[9];
  const float* l2W   = (const float*)d_in[10];
  const float* l2b   = (const float*)d_in[11];
  const float* l3W   = (const float*)d_in[12];
  const float* ltW   = (const float*)d_in[13];
  const float* ltb   = (const float*)d_in[14];
  float* out = (float*)d_out;

  // workspace layout (aliased by lifetime): ~8.3 MB, frag arrays 32B-aligned
  float*  h      = (float*)d_ws;                         // [B,L,H]
  float*  mulseq = h + (size_t)kB * kL * kH;             // [B,L,NH*H]
  float*  mulone = h;                                    // reuses h (dead after 1st GAT)
  float*  seqhid = mulseq;                               // reuses mulseq (dead after k_mulone)
  float*  afin   = mulseq + (size_t)kB * kL * kNH * kH;  // [B,H]
  int*    seql   = (int*)(afin + (size_t)kB * kH);       // [B] (512 B)
  __bf16* ahfrag = (__bf16*)(seql + 128);                // 8192 bf16 (16 KB)
  __bf16* alfrag = ahfrag + 256 * 2 * 16;                // 8192 bf16 (16 KB)

  k_gather<<<(kB * kL * kH + 255) / 256, 256, 0, stream>>>(path, emb, h);
  k_seqlen<<<1, kB, 0, stream>>>(mask, seql);
  k_gat<<<kNH * kB, 256, 0, stream>>>(h, W_in, a_in, mask, mulseq, kNH, kNH * kH, 1);
  k_mulone<<<(kB * kL * kH + 255) / 256, 256, 0, stream>>>(mulseq, w, mulone);
  k_gat<<<kB, 256, 0, stream>>>(mulone, W_out, a_out, mask, seqhid, 1, kH, 0);
  k_readout<<<kB, kH, 0, stream>>>(seqhid, seql, mask, l1W, l1b, l2W, l2b, l3W, ltW, ltb, afin);
  k_afrag<<<1, 256, 0, stream>>>(afin, ahfrag, alfrag);
  k_scores<<<kNU / 16, 256, 0, stream>>>(ahfrag, alfrag, emb, out);
}
